// BidirectionalConvLSTM_70282844832225
// MI455X (gfx1250) — matrix operand order repaired and checked by reading, not compiled, not run on hardware
//
#include <hip/hip_runtime.h>
#include <hip/hip_bf16.h>
#include <math.h>

// ---------------------------------------------------------------------------
// Bidirectional ConvLSTM on MI455X (gfx1250, wave32).
// Conv steps lowered to implicit GEMM on V_WMMA_F32_16X16X32_BF16.
// Sizes fixed by setup_inputs(): B=4, T=16, C=64, HID=64, H=W=64, k=3,
// GROUPS=4 (group == gate block of 64 channels).
//
// K ordering for the conv GEMM:  k = (ky*3+kx)*128 + c   (kt-major, c-minor)
// so every 32-wide K slab has scalar (ky,kx) and a scalar x-vs-h source.
// ---------------------------------------------------------------------------

#define T_STEPS 16
#define BATCH   4
#define HID     64
#define CIN     64
#define KC      128            // CIN + HID
#define G4      256            // 4*HID gate channels
#define HW      4096           // 64*64
#define KK      1152           // 9 * KC (im2col K)
#define M_CONV  16384          // BATCH * HW
#define M_FUSE  262144         // T*B*HW
#define EPS     1e-5f

typedef __attribute__((ext_vector_type(16))) __bf16 v16bf;
typedef __attribute__((ext_vector_type(8)))  float  v8f;

// Fragment assembled from two explicit 128-bit LDS loads (ds_load_b128).
union BF16Frag { uint4 q[2]; unsigned int u[8]; v16bf v; };

__device__ __forceinline__ unsigned short f2bf(float f) {
    unsigned int u = __float_as_uint(f);
    u += 0x7FFFu + ((u >> 16) & 1u);          // round-to-nearest-even
    return (unsigned short)(u >> 16);
}

// ---------------------------------------------------------------------------
// Weight packing: Wconv [256][128][3][3] f32 -> bf16 [K=1152][N=256] with
// k = (ky*3+kx)*128 + c.
// ---------------------------------------------------------------------------
__global__ void pack_conv_w(const float* __restrict__ W, unsigned short* __restrict__ wpk) {
    int i = blockIdx.x * 256 + threadIdx.x;        // 1152*256
    if (i >= KK * G4) return;
    int k = i >> 8, oc = i & 255;
    int kt = k >> 7, c = k & 127;                  // kt = ky*3+kx
    int ky = kt / 3, kx = kt - ky * 3;
    wpk[(size_t)k * G4 + oc] = f2bf(W[((oc * KC + c) * 3 + ky) * 3 + kx]);
}

__global__ void pack_fuse_w(const float* __restrict__ W, unsigned short* __restrict__ wpk) {
    int i = blockIdx.x * 256 + threadIdx.x;        // 128*64
    if (i >= KC * HID) return;
    int k = i >> 6, oc = i & 63;
    wpk[k * HID + oc] = f2bf(W[oc * KC + k]);      // Wfu is [64][128][1][1]
}

__global__ void zero_f32(float* __restrict__ p, int n) {
    int i = blockIdx.x * 256 + threadIdx.x;
    if (i < n) p[i] = 0.f;
}

// ---------------------------------------------------------------------------
// Implicit-GEMM conv step:  gates = conv3x3(concat(x_t, h_prev), W)
//   M = 16384 pixels, N = 256, K = 1152.  Block tile 128(M) x 64(N), 8 waves.
// GroupNorm partial sums fused in the epilogue (tile lies in one (b, group)).
// ---------------------------------------------------------------------------
__global__ __launch_bounds__(256)
void conv_step(const float* __restrict__ x, const float* __restrict__ hprev,
               const unsigned short* __restrict__ wpk,
               float* __restrict__ gates, float* __restrict__ gstats, int t)
{
    // Row stride 40 ushorts = 80 B: multiple of 16 so uint4 fragment loads
    // are 16B-aligned -> ds_load_b128.
    __shared__ __align__(16) unsigned short As[128][40];
    __shared__ __align__(16) unsigned short Bs[64][40];
    __shared__ float Ds[128][66];      // f32 D staging, conflict-free stride
    __shared__ float redS[2];

    const int tid  = threadIdx.x;
    const int wave = tid >> 5, lane = tid & 31;
    const int hlf  = lane >> 4, l16 = lane & 15;
    const int mtile = blockIdx.x * 128;
    const int ntile = blockIdx.y * 64;
    const int b    = mtile >> 12;          // whole M tile inside one batch
    const int hw0  = mtile & 4095;

    if (tid < 2) redS[tid] = 0.f;          // visible after first K-loop barrier

    const float* xplane = x + (((size_t)(b * T_STEPS + t)) << 18);  // [64][4096]
    const float* hplane = hprev + ((size_t)b << 18);                // [64][4096]

    v8f acc[4] = {};

    for (int k0 = 0; k0 < KK; k0 += 32) {
        // ---- scalar slab parameters ----
        const int kt = k0 >> 7;            // ky*3+kx, uniform
        const int c0 = k0 & 127;           // channel base, uniform
        const int dy = kt / 3 - 1;
        const int dx = kt - (kt / 3) * 3 - 1;
        const float* src = (c0 < CIN) ? (xplane + ((size_t)c0 << 12))
                                      : (hplane + ((size_t)(c0 - CIN) << 12));

        // ---- A tile: coalesced im2col gather (rows = pixels, fastest) ----
        for (int i = tid; i < 128 * 32; i += 256) {
            int row = i & 127, kk = i >> 7;
            int hw = hw0 + row;
            int y  = hw >> 6, xc = hw & 63;
            int yy = y + dy, xx = xc + dx;
            float v = 0.f;
            if ((unsigned)yy < 64u && (unsigned)xx < 64u)
                v = src[((size_t)kk << 12) + (yy << 6) + xx];
            As[row][kk] = f2bf(v);
        }
        // ---- B tile: pre-packed bf16 weights, stored [n][k] ----
        for (int i = tid; i < 64 * 32; i += 256) {
            int n = i >> 5, kk = i & 31;
            Bs[n][kk] = wpk[(size_t)(k0 + kk) * G4 + ntile + n];
        }
        if (k0 + 32 < KK)   // global_prefetch_b8 on next weight slab
            __builtin_prefetch(&wpk[(size_t)(k0 + 32) * G4 + ntile], 0, 1);
        __syncthreads();

        // ---- fragments (explicit 128-bit LDS loads) + WMMA ----
        BF16Frag a;
        {
            int m = wave * 16 + l16;
            a.q[0] = *(const uint4*)&As[m][hlf * 8];        // K pairs j=0..3
            a.q[1] = *(const uint4*)&As[m][16 + hlf * 8];   // K pairs j=4..7
        }
        #pragma unroll
        for (int nf = 0; nf < 4; ++nf) {
            BF16Frag bb;
            int n = nf * 16 + l16;
            bb.q[0] = *(const uint4*)&Bs[n][hlf * 8];       // K pairs j=0..3
            bb.q[1] = *(const uint4*)&Bs[n][16 + hlf * 8];  // K pairs j=4..7
            acc[nf] = __builtin_amdgcn_wmma_f32_16x16x32_bf16(
                false, a.v, false, bb.v, (short)0, acc[nf], false, false);
        }
        __syncthreads();
    }

    // ---- epilogue: GN partial sums + LDS-staged coalesced stores ----
    float s = 0.f, sq = 0.f;
    #pragma unroll
    for (int nf = 0; nf < 4; ++nf) {
        #pragma unroll
        for (int r = 0; r < 8; ++r) {
            float v = acc[nf][r];
            Ds[wave * 16 + r + hlf * 8][nf * 16 + l16] = v;
            s += v; sq += v * v;
        }
    }
    atomicAdd(&redS[0], s);
    atomicAdd(&redS[1], sq);
    __syncthreads();
    for (int i = tid; i < 128 * 64; i += 256) {      // hw fastest: coalesced
        int row = i & 127, col = i >> 7;
        gates[(((size_t)(b << 8) + ntile + col) << 12) + hw0 + row] = Ds[row][col];
    }
    if (tid == 0) {
        int g = ntile >> 6;                           // tile in one group
        atomicAdd(&gstats[(b * 4 + g) * 2 + 0], redS[0]);
        atomicAdd(&gstats[(b * 4 + g) * 2 + 1], redS[1]);
    }
}

// ---------------------------------------------------------------------------
// GroupNorm affine + gates + LSTM state update.  One thread per (b,ch,hw).
// Writes h (f32 state for next conv) and hs slab (bf16 for fusion GEMM).
// ---------------------------------------------------------------------------
__global__ __launch_bounds__(256)
void gn_lstm_apply(const float* __restrict__ gates, const float* __restrict__ gstats,
                   const float* __restrict__ gw, const float* __restrict__ gb,
                   float* __restrict__ h, float* __restrict__ c,
                   unsigned short* __restrict__ hs_t)
{
    int idx = blockIdx.x * 256 + threadIdx.x;      // B*64*4096 = 1,048,576
    int b   = idx >> 18;
    int rem = idx & 262143;
    int ch  = rem >> 12;
    int hw  = rem & 4095;
    const float invN = 1.f / 262144.f;             // (C/G)*H*W per (b,group)
    float vals[4];
    #pragma unroll
    for (int q = 0; q < 4; ++q) {
        int cc    = q * HID + ch;
        float mu  = gstats[(b * 4 + q) * 2 + 0] * invN;
        float var = gstats[(b * 4 + q) * 2 + 1] * invN - mu * mu;
        float inv = rsqrtf(var + EPS);
        float v = gates[((((size_t)b << 8) + cc) << 12) + hw];
        vals[q] = (v - mu) * inv * gw[cc] + gb[cc];
    }
    float ig = 1.f / (1.f + __expf(-vals[0]));
    float fg = 1.f / (1.f + __expf(-vals[1]));
    float og = 1.f / (1.f + __expf(-vals[2]));
    float gg = tanhf(vals[3]);
    float cn = fg * c[idx] + ig * gg;
    float hn = og * tanhf(cn);
    c[idx] = cn;
    h[idx] = hn;
    hs_t[idx] = f2bf(hn);
}

// ---------------------------------------------------------------------------
// Fusion 1x1 conv:  out(pre-BN) = concat(hf,hb) @ Wfu^T
//   M = 262144, N = 64, K = 128.  hf/hb already bf16.  BN sums fused.
// ---------------------------------------------------------------------------
__global__ __launch_bounds__(256)
void fuse_gemm(const unsigned short* __restrict__ hf, const unsigned short* __restrict__ hb,
               const unsigned short* __restrict__ wpk,
               float* __restrict__ out, float* __restrict__ bstats)
{
    __shared__ __align__(16) unsigned short As[128][40];
    __shared__ __align__(16) unsigned short Bs[64][40];
    __shared__ float Ds[128][66];
    __shared__ float chS[128];

    const int tid  = threadIdx.x;
    const int wave = tid >> 5, lane = tid & 31;
    const int hlf  = lane >> 4, l16 = lane & 15;
    const int mtile = blockIdx.x * 128;
    const int tb   = mtile >> 12;          // t*B + b, scalar for the tile
    const int hw0  = mtile & 4095;
    const int tt   = tb >> 2, bb_ = tb & 3;

    v8f acc[4] = {};

    for (int k0 = 0; k0 < KC; k0 += 32) {
        const unsigned short* src = (k0 < HID)
            ? (hf + (((size_t)tb * HID + k0) << 12))
            : (hb + (((size_t)tb * HID + (k0 - HID)) << 12));
        for (int i = tid; i < 128 * 32; i += 256) {   // rows fastest: coalesced
            int row = i & 127, kk = i >> 7;
            As[row][kk] = src[((size_t)kk << 12) + hw0 + row];
        }
        for (int i = tid; i < 64 * 32; i += 256) {
            int n = i >> 5, kk = i & 31;
            Bs[n][kk] = wpk[(k0 + kk) * HID + n];
        }
        __syncthreads();

        BF16Frag a;
        int m = wave * 16 + l16;
        a.q[0] = *(const uint4*)&As[m][hlf * 8];
        a.q[1] = *(const uint4*)&As[m][16 + hlf * 8];
        #pragma unroll
        for (int nf = 0; nf < 4; ++nf) {
            BF16Frag bfr;
            int n = nf * 16 + l16;
            bfr.q[0] = *(const uint4*)&Bs[n][hlf * 8];
            bfr.q[1] = *(const uint4*)&Bs[n][16 + hlf * 8];
            acc[nf] = __builtin_amdgcn_wmma_f32_16x16x32_bf16(
                false, a.v, false, bfr.v, (short)0, acc[nf], false, false);
        }
        __syncthreads();
    }

    // ---- epilogue: BN channel sums + LDS-staged coalesced stores ----
    for (int i = tid; i < 128; i += 256) chS[i] = 0.f;
    __syncthreads();
    #pragma unroll
    for (int nf = 0; nf < 4; ++nf) {
        int oc = nf * 16 + l16;
        float s = 0.f, sq = 0.f;
        #pragma unroll
        for (int r = 0; r < 8; ++r) {
            float v = acc[nf][r];
            Ds[wave * 16 + r + hlf * 8][oc] = v;
            s += v; sq += v * v;
        }
        atomicAdd(&chS[oc * 2 + 0], s);
        atomicAdd(&chS[oc * 2 + 1], sq);
    }
    __syncthreads();
    for (int i = tid; i < 128 * 64; i += 256) {       // hw fastest: coalesced
        int row = i & 127, col = i >> 7;
        out[(((size_t)(bb_ * T_STEPS + tt) << 6) + col) * 4096 + hw0 + row] = Ds[row][col];
    }
    for (int i = tid; i < 64; i += 256) {
        atomicAdd(&bstats[i * 2 + 0], chS[i * 2 + 0]);
        atomicAdd(&bstats[i * 2 + 1], chS[i * 2 + 1]);
    }
}

// BatchNorm (batch stats) + ReLU, in place on d_out [B][T][64][H][W].
__global__ __launch_bounds__(256)
void bn_relu(float* __restrict__ out, const float* __restrict__ bstats,
             const float* __restrict__ bw, const float* __restrict__ bbias)
{
    int idx = blockIdx.x * 256 + threadIdx.x;      // 16,777,216
    int oc  = (idx >> 12) & 63;
    const float invN = 1.f / 262144.f;             // T*B*H*W per channel
    float mu  = bstats[oc * 2 + 0] * invN;
    float var = bstats[oc * 2 + 1] * invN - mu * mu;
    float v = (out[idx] - mu) * rsqrtf(var + EPS) * bw[oc] + bbias[oc];
    out[idx] = fmaxf(v, 0.f);
}

// ---------------------------------------------------------------------------
// Launcher.  Workspace (~120 MB):
//   f32: gates_f gates_b [4M ea] | hF cF hB cB [1M ea] | stats [192]
//   bf16: hf hb [16M ea] | packed conv weights [294912 ea] | fuse w [8192]
// ---------------------------------------------------------------------------
extern "C" void kernel_launch(void* const* d_in, const int* in_sizes, int n_in,
                              void* d_out, int out_size, void* d_ws, size_t ws_size,
                              hipStream_t stream) {
    const float* x     = (const float*)d_in[0];
    const float* Wf    = (const float*)d_in[1];
    const float* gnf_w = (const float*)d_in[2];
    const float* gnf_b = (const float*)d_in[3];
    const float* Wb    = (const float*)d_in[4];
    const float* gnb_w = (const float*)d_in[5];
    const float* gnb_b = (const float*)d_in[6];
    const float* Wfu   = (const float*)d_in[7];
    const float* bn_w  = (const float*)d_in[8];
    const float* bn_b  = (const float*)d_in[9];
    float* out = (float*)d_out;

    const size_t SLAB = (size_t)BATCH * HID * HW;          // 1,048,576
    float* ws       = (float*)d_ws;
    float* gates_f  = ws;                                  // 4,194,304
    float* gates_b  = gates_f + (size_t)BATCH * G4 * HW;   // 4,194,304
    float* hF       = gates_b + (size_t)BATCH * G4 * HW;
    float* cF       = hF + SLAB;
    float* hB       = cF + SLAB;
    float* cB       = hB + SLAB;
    float* gstats_f = cB + SLAB;                           // 32
    float* gstats_b = gstats_f + 32;                       // 32
    float* bnstats  = gstats_b + 32;                       // 128
    unsigned short* hf_bf = (unsigned short*)(bnstats + 128);  // 16,777,216
    unsigned short* hb_bf = hf_bf + (size_t)T_STEPS * SLAB;    // 16,777,216
    unsigned short* wpk_f = hb_bf + (size_t)T_STEPS * SLAB;    // 294,912
    unsigned short* wpk_b = wpk_f + (size_t)KK * G4;
    unsigned short* wpk_u = wpk_b + (size_t)KK * G4;           // 8,192

    // Pack weights to bf16 [K][N].
    pack_conv_w<<<(KK * G4 + 255) / 256, 256, 0, stream>>>(Wf, wpk_f);
    pack_conv_w<<<(KK * G4 + 255) / 256, 256, 0, stream>>>(Wb, wpk_b);
    pack_fuse_w<<<(KC * HID + 255) / 256, 256, 0, stream>>>(Wfu, wpk_u);

    // Zero h/c states (hF..cB contiguous) and BN stats.
    zero_f32<<<(int)((4 * SLAB + 255) / 256), 256, 0, stream>>>(hF, (int)(4 * SLAB));
    zero_f32<<<1, 128, 0, stream>>>(bnstats, 128);

    dim3 cgrid(M_CONV / 128, G4 / 64);   // 128 x 4 tiles

    // Forward scan.
    for (int t = 0; t < T_STEPS; ++t) {
        zero_f32<<<1, 32, 0, stream>>>(gstats_f, 32);
        conv_step<<<cgrid, 256, 0, stream>>>(x, hF, wpk_f, gates_f, gstats_f, t);
        gn_lstm_apply<<<(int)(SLAB / 256), 256, 0, stream>>>(
            gates_f, gstats_f, gnf_w, gnf_b, hF, cF, hf_bf + (size_t)t * SLAB);
    }
    // Backward scan.
    for (int s = 0; s < T_STEPS; ++s) {
        int t = T_STEPS - 1 - s;
        zero_f32<<<1, 32, 0, stream>>>(gstats_b, 32);
        conv_step<<<cgrid, 256, 0, stream>>>(x, hB, wpk_b, gates_b, gstats_b, t);
        gn_lstm_apply<<<(int)(SLAB / 256), 256, 0, stream>>>(
            gates_b, gstats_b, gnb_w, gnb_b, hB, cB, hb_bf + (size_t)t * SLAB);
    }

    // Fusion 1x1 conv (WMMA) -> pre-BN d_out, with fused BN sums.
    fuse_gemm<<<M_FUSE / 128, 256, 0, stream>>>(hf_bf, hb_bf, wpk_u, out, bnstats);
    // BN (batch stats) + ReLU in place.
    bn_relu<<<(out_size + 255) / 256, 256, 0, stream>>>(out, bnstats, bn_w, bn_b);
}
